// DecoderLSTM_68736656605742
// MI455X (gfx1250) — matrix operand order repaired and checked by reading, not compiled, not run on hardware
//
#include <hip/hip_runtime.h>
#include <hip/hip_bf16.h>

// ---------------------------------------------------------------------------
// DecoderLSTM on gfx1250 (MI455X): fp16 WMMA (v_wmma_f32_16x16x32_f16) GEMMs
// with fp32 accumulation, fused LSTM pointwise via LDS, fp32 cell state.
// Register blocking: 2 M-tiles per wave so every B fragment feeds 2 WMMAs
// (halves W_lin / W_ih / W_hh L2 traffic, which is the binding resource).
// ---------------------------------------------------------------------------

typedef _Float16 half_t;
typedef __attribute__((ext_vector_type(16))) _Float16 v16h;
typedef __attribute__((ext_vector_type(8)))  _Float16 v8h;
typedef __attribute__((ext_vector_type(8)))  float    v8f;

#define B_    256
#define T_    32
#define V_    10000
#define E_    512
#define H_    512
#define FOURH 2048

// ---- WMMA helpers ---------------------------------------------------------

__device__ __forceinline__ v8f wmma_f16(v16h a, v16h b, v8f c) {
  // D = A(16x32 f16) * B(32x16 f16) + C(16x16 f32)
  return __builtin_amdgcn_wmma_f32_16x16x32_f16(
      /*neg_a=*/false, a, /*neg_b=*/false, b,
      /*c_mod=*/(short)0, c, /*reuse_a=*/false, /*reuse_b=*/false);
}

// A fragment 16x32 (f16): rowp_k0 = pointer to this lane's row (row = lane&15)
// already offset to k0. ISA layout: lane<16 holds K {0..7,16..23}, lane>=16
// holds K {8..15,24..31}; element e<8 -> K = ksel*8+e, e>=8 -> K = 16+ksel*8+(e-8).
__device__ __forceinline__ v16h load_a_row(const half_t* __restrict__ rowp_k0,
                                           int lane) {
  const int ksel8 = (lane >> 4) * 8;
  const half_t* p = rowp_k0 + ksel8;
  v8h lo = *(const v8h*)(p);        // K = ksel8 .. ksel8+7
  v8h hi = *(const v8h*)(p + 16);   // K = 16+ksel8 .. 16+ksel8+7
  v16h a;
#pragma unroll
  for (int i = 0; i < 8; ++i) { a[i] = lo[i]; a[i + 8] = hi[i]; }
  return a;
}

// B fragment 32x16 (f16) from row-major weight W[N x K] (so B = W^T tile):
// base = &W[n0*ld + k0]. Lane holds column n = lane&15; lanes 0-15 carry
// K {0..7,16..23}, lanes 16-31 carry K {8..15,24..31}; element e<8 -> K = (lane>>4)*8 + e, e>=8 -> K = 16+(lane>>4)*8+(e-8).
__device__ __forceinline__ v16h load_b_frag(const half_t* __restrict__ base,
                                            int ld, int lane) {
  const int n    = lane & 15;
  const int koff = (lane >> 4) << 3;  // 0 or 8
  const half_t* p = base + (size_t)n * ld + koff;
  v8h lo = *(const v8h*)(p);
  v8h hi = *(const v8h*)(p + 16);
  v16h b;
#pragma unroll
  for (int i = 0; i < 8; ++i) { b[i] = lo[i]; b[i + 8] = hi[i]; }
  return b;
}

__device__ __forceinline__ float sigf(float x) {
  return 1.0f / (1.0f + __expf(-x));
}

// ---- Fused gates GEMM + LSTM cell update ----------------------------------
// Block = 128 threads = 4 waves; wave w computes gate w's 32x16 slab (two
// 16x16 M-tiles, B fragment reused for both) as x@W_ih^T + h@W_hh^T over
// K=512+512. The 4 gate slabs meet in LDS for the pointwise cell update.
// h ping-pongs (h_prev -> h_next); c is tile-private and updates in place.
__global__ __launch_bounds__(128)
void lstm_gates_kernel(const int* __restrict__ caps, int tstep,
                       const half_t* __restrict__ embed16,
                       const half_t* __restrict__ Wih16,
                       const half_t* __restrict__ Whh16,
                       const float*  __restrict__ bias,     // b_ih + b_hh [4H]
                       const half_t* __restrict__ h_prev,
                       half_t*       __restrict__ h_next,
                       float*        __restrict__ c_state) {
  __shared__ float gtile[4][32][16];   // [gate][row in 32-slab][col]

  const int lane = threadIdx.x & 31;
  const int wave = threadIdx.x >> 5;          // gate index: 0=i 1=f 2=g 3=o
  const int m0   = blockIdx.y * 32;           // 32 batch rows (2 M-tiles)
  const int j0   = blockIdx.x * 16;           // hidden tile
  const int n0   = wave * H_ + j0;            // column block in [4H]

  // Embedding gather: lane's A-rows are embed16[caps[m0(+16)+m, tstep-1]].
  const int m    = lane & 15;
  const int tok0 = caps[(m0 + m) * T_ + (tstep - 1)];
  const int tok1 = caps[(m0 + 16 + m) * T_ + (tstep - 1)];
  const half_t* xrow0 = embed16 + (size_t)tok0 * E_;
  const half_t* xrow1 = embed16 + (size_t)tok1 * E_;
  const half_t* hrow0 = h_prev + (size_t)(m0 + m) * H_;
  const half_t* hrow1 = hrow0 + (size_t)16 * H_;

  v8f acc0 = {};
  v8f acc1 = {};
#pragma unroll 4
  for (int k0 = 0; k0 < E_; k0 += 32) {
    v16h a0 = load_a_row(xrow0 + k0, lane);
    v16h a1 = load_a_row(xrow1 + k0, lane);
    v16h b  = load_b_frag(Wih16 + (size_t)n0 * E_ + k0, E_, lane);
    acc0 = wmma_f16(a0, b, acc0);
    acc1 = wmma_f16(a1, b, acc1);
  }
#pragma unroll 4
  for (int k0 = 0; k0 < H_; k0 += 32) {
    v16h a0 = load_a_row(hrow0 + k0, lane);
    v16h a1 = load_a_row(hrow1 + k0, lane);
    v16h b  = load_b_frag(Whh16 + (size_t)n0 * H_ + k0, H_, lane);
    acc0 = wmma_f16(a0, b, acc0);
    acc1 = wmma_f16(a1, b, acc1);
  }

  // D layout: element r -> row (lane>>4)*8 + r, col lane&15
  const int rbase = (lane >> 4) * 8;
  const int cix0  = lane & 15;
#pragma unroll
  for (int r = 0; r < 8; ++r) {
    gtile[wave][rbase + r][cix0]      = acc0[r];
    gtile[wave][16 + rbase + r][cix0] = acc1[r];
  }
  __syncthreads();

  // Pointwise LSTM over the 32x16 slab (512 elems, 128 threads -> 4 each).
#pragma unroll
  for (int e = threadIdx.x; e < 512; e += 128) {
    const int r   = e >> 4;            // 0..31
    const int cix = e & 15;
    const int col = j0 + cix;
    const float ig = gtile[0][r][cix] + bias[0 * H_ + col];
    const float fg = gtile[1][r][cix] + bias[1 * H_ + col];
    const float gg = gtile[2][r][cix] + bias[2 * H_ + col];
    const float og = gtile[3][r][cix] + bias[3 * H_ + col];
    const size_t idx = (size_t)(m0 + r) * H_ + col;
    const float cn = sigf(fg) * c_state[idx] + sigf(ig) * tanhf(gg);
    c_state[idx] = cn;
    h_next[idx]  = (half_t)(sigf(og) * tanhf(cn));
  }
}

// ---- Logits GEMM: out[t] = h @ W_lin^T + b_lin ----------------------------
// Block = 160 threads = 5 waves; each wave owns 2 M-tiles x 5 N-tiles
// (10 accumulators). Every B fragment is loaded once per K-slab and feeds
// two WMMAs; the A fragments are reused across the 5 N-tiles.
// 625 N-tiles = grid.x(25) * 5 waves * 5; 16 M-tiles = grid.y(8) * 2.
__global__ __launch_bounds__(160)
void logits_kernel(const half_t* __restrict__ h16,
                   const half_t* __restrict__ Wlin16,
                   const float*  __restrict__ b_lin,
                   float*        __restrict__ out_t) {
  const int lane = threadIdx.x & 31;
  const int wave = threadIdx.x >> 5;                 // 0..4
  const int m0   = blockIdx.y * 32;                  // 2 M-tiles: m0, m0+16
  const int ntile_base = blockIdx.x * 25 + wave * 5; // first of 5 N-tiles

  const half_t* hrow0 = h16 + (size_t)(m0 + (lane & 15)) * H_;
  const half_t* hrow1 = hrow0 + (size_t)16 * H_;

  const v8f vzero = {};
  v8f acc0[5], acc1[5];
#pragma unroll
  for (int j = 0; j < 5; ++j) { acc0[j] = vzero; acc1[j] = vzero; }

#pragma unroll 2
  for (int k0 = 0; k0 < H_; k0 += 32) {
    v16h a0 = load_a_row(hrow0 + k0, lane);
    v16h a1 = load_a_row(hrow1 + k0, lane);
#pragma unroll
    for (int j = 0; j < 5; ++j) {
      const int n0 = (ntile_base + j) * 16;
      v16h b = load_b_frag(Wlin16 + (size_t)n0 * H_ + k0, H_, lane);
      acc0[j] = wmma_f16(a0, b, acc0[j]);
      acc1[j] = wmma_f16(a1, b, acc1[j]);
    }
  }

  const int rbase = (lane >> 4) * 8;
#pragma unroll
  for (int j = 0; j < 5; ++j) {
    const int col = (ntile_base + j) * 16 + (lane & 15);
    const float bl = b_lin[col];
#pragma unroll
    for (int r = 0; r < 8; ++r) {
      out_t[(size_t)(m0 + rbase + r) * V_ + col]      = acc0[j][r] + bl;
      out_t[(size_t)(m0 + 16 + rbase + r) * V_ + col] = acc1[j][r] + bl;
    }
  }
}

// ---- Utility kernels ------------------------------------------------------

__global__ void f32_to_f16_kernel(const float* __restrict__ in,
                                  half_t* __restrict__ out, int n) {
  const int i = blockIdx.x * blockDim.x + threadIdx.x;
  if (i < n) out[i] = (half_t)in[i];
}

__global__ void zero_f32_kernel(float* __restrict__ p, int n) {
  const int i = blockIdx.x * blockDim.x + threadIdx.x;
  if (i < n) p[i] = 0.0f;
}

__global__ void bias_sum_kernel(const float* __restrict__ a,
                                const float* __restrict__ b,
                                float* __restrict__ o, int n) {
  const int i = blockIdx.x * blockDim.x + threadIdx.x;
  if (i < n) o[i] = a[i] + b[i];
}

// ---- Launch ---------------------------------------------------------------

extern "C" void kernel_launch(void* const* d_in, const int* in_sizes, int n_in,
                              void* d_out, int out_size, void* d_ws, size_t ws_size,
                              hipStream_t stream) {
  (void)in_sizes; (void)n_in; (void)out_size; (void)ws_size;

  const int*   caps   = (const int*)  d_in[0];   // [B,T]
  const float* latent = (const float*)d_in[1];   // [B,H]
  const float* embed  = (const float*)d_in[2];   // [V,E]
  const float* W_ih   = (const float*)d_in[3];   // [4H,E]
  const float* W_hh   = (const float*)d_in[4];   // [4H,H]
  const float* b_ih   = (const float*)d_in[5];   // [4H]
  const float* b_hh   = (const float*)d_in[6];   // [4H]
  const float* W_lin  = (const float*)d_in[7];   // [V,H]
  const float* b_lin  = (const float*)d_in[8];   // [V]
  float* out = (float*)d_out;                    // [T,B,V]

  // Workspace carve-out (256B aligned regions), ~25.8 MB total.
  char*  ws  = (char*)d_ws;
  size_t off = 0;
  auto carve = [&](size_t bytes) -> void* {
    off = (off + 255) & ~(size_t)255;
    void* p = ws + off;
    off += bytes;
    return p;
  };
  half_t* embed16 = (half_t*)carve((size_t)V_ * E_ * sizeof(half_t));
  half_t* Wih16   = (half_t*)carve((size_t)FOURH * E_ * sizeof(half_t));
  half_t* Whh16   = (half_t*)carve((size_t)FOURH * H_ * sizeof(half_t));
  half_t* Wlin16  = (half_t*)carve((size_t)V_ * H_ * sizeof(half_t));
  half_t* h16     = (half_t*)carve((size_t)2 * B_ * H_ * sizeof(half_t)); // ping-pong
  float*  c_ws    = (float*) carve((size_t)B_ * H_ * sizeof(float));
  float*  bias_ws = (float*) carve((size_t)FOURH * sizeof(float));

  const int THR = 256;
  auto blocks = [&](int n) { return (n + THR - 1) / THR; };

  // One-time (per call) conversions / initialization — weights become
  // fp16 and L2-resident for the 31 sequential steps.
  f32_to_f16_kernel<<<blocks(V_ * E_),     THR, 0, stream>>>(embed, embed16, V_ * E_);
  f32_to_f16_kernel<<<blocks(FOURH * E_),  THR, 0, stream>>>(W_ih,  Wih16,  FOURH * E_);
  f32_to_f16_kernel<<<blocks(FOURH * H_),  THR, 0, stream>>>(W_hh,  Whh16,  FOURH * H_);
  f32_to_f16_kernel<<<blocks(V_ * H_),     THR, 0, stream>>>(W_lin, Wlin16, V_ * H_);
  f32_to_f16_kernel<<<blocks(B_ * H_),     THR, 0, stream>>>(latent, h16,   B_ * H_); // h0 -> buf0
  bias_sum_kernel  <<<blocks(FOURH),       THR, 0, stream>>>(b_ih, b_hh, bias_ws, FOURH);
  zero_f32_kernel  <<<blocks(B_ * H_),     THR, 0, stream>>>(c_ws, B_ * H_);          // c0 = 0
  zero_f32_kernel  <<<blocks(B_ * V_),     THR, 0, stream>>>(out,  B_ * V_);          // outputs[0] = 0

  // Sequential teacher-forced steps t = 1..T-1.
  for (int t = 1; t < T_; ++t) {
    const half_t* hp = h16 + (size_t)((t - 1) & 1) * B_ * H_;
    half_t*       hn = h16 + (size_t)(t & 1) * B_ * H_;

    lstm_gates_kernel<<<dim3(H_ / 16, B_ / 32), 128, 0, stream>>>(
        caps, t, embed16, Wih16, Whh16, bias_ws, hp, hn, c_ws);

    logits_kernel<<<dim3(25, B_ / 32), 160, 0, stream>>>(
        hn, Wlin16, b_lin, out + (size_t)t * B_ * V_);
  }
}
